// MultiHeadAttention_51196010168886
// MI455X (gfx1250) — compile-verified, matrix operand order checked by reading, not run on hardware
//
#include <hip/hip_runtime.h>
#include <hip/hip_bf16.h>

// MI455X / gfx1250: wave32, WMMA 16x16x32 f16 -> f32.
// MHA forward: QKV proj (WMMA GEMM, async-to-LDS staged, double buffered)
// -> flash attention (WMMA, async K staging) -> out proj (WMMA GEMM + bias +
// residual) -> LayerNorm. All f16 matmul inputs, f32 accumulate.

typedef __attribute__((ext_vector_type(16))) _Float16 v16h;
typedef __attribute__((ext_vector_type(8)))  float    v8f;

#define B_  2
#define S_  2048
#define D_  1024
#define H_  16
#define HD_ 64

// ---------------------------------------------------------------------------
// Fragment helpers (16-bit A/B layout per CDNA5 ISA 7.12.2):
// lane L holds row (L&15); chunk0 = K[(L>>4)*8 .. +8), chunk1 = K[16+(L>>4)*8 .. +8)
// Works for row-major A (MxK) and N-major B (NxK) buffers alike.
// ---------------------------------------------------------------------------
__device__ __forceinline__ v16h ldfrag(const _Float16* base, int ld, int lane) {
  const int r  = lane & 15;
  const int hf = lane >> 4;
  union { v16h f; uint4 q[2]; } u;
  u.q[0] = *(const uint4*)(base + r * ld + hf * 8);
  u.q[1] = *(const uint4*)(base + r * ld + 16 + hf * 8);
  return u.f;
}

__device__ __forceinline__ v8f wmma_f16(v16h a, v16h b, v8f c) {
  return __builtin_amdgcn_wmma_f32_16x16x32_f16(false, a, false, b, (short)0, c,
                                                false, false);
}

// Async global->LDS copy of 32B (2 x B128); INST_OFFSET applies to both sides.
__device__ __forceinline__ void async_copy32(unsigned lds_addr, const void* gptr) {
  const unsigned long long ga = (unsigned long long)(uintptr_t)gptr;
  asm volatile("global_load_async_to_lds_b128 %0, %1, off"
               :: "v"(lds_addr), "v"(ga) : "memory");
  asm volatile("global_load_async_to_lds_b128 %0, %1, off offset:16"
               :: "v"(lds_addr), "v"(ga) : "memory");
}

__device__ __forceinline__ unsigned lds_off(const void* p) {
  return (unsigned)(uintptr_t)p;   // generic LDS pointer: low 32 bits = LDS addr
}

// ---------------------------------------------------------------------------
// x (f32) -> f16, 4 elements/thread
// ---------------------------------------------------------------------------
__global__ __launch_bounds__(256) void cvt_f16_kernel(const float* __restrict__ in,
                                                      _Float16* __restrict__ out) {
  const int i = (blockIdx.x * 256 + threadIdx.x) * 4;
  float4 v = *(const float4*)(in + i);
  union { _Float16 h[4]; uint2 u; } o;
  o.h[0] = (_Float16)v.x; o.h[1] = (_Float16)v.y;
  o.h[2] = (_Float16)v.z; o.h[3] = (_Float16)v.w;
  *(uint2*)(out + i) = o.u;
}

// ---------------------------------------------------------------------------
// W (f32, [K][N] row-major) -> WT (f16, [N][K] N-major) via LDS tile transpose
// ---------------------------------------------------------------------------
__global__ __launch_bounds__(256) void transpose_cvt_kernel(const float* __restrict__ W,
                                                            _Float16* __restrict__ WT) {
  __shared__ float tile[32][33];
  const int tx = threadIdx.x, ty = threadIdx.y;
  const int bk = blockIdx.x * 32;   // K tile origin
  const int bn = blockIdx.y * 32;   // N tile origin
#pragma unroll
  for (int i = 0; i < 4; ++i)
    tile[ty + i * 8][tx] = W[(bk + ty + i * 8) * D_ + bn + tx];
  __syncthreads();
#pragma unroll
  for (int i = 0; i < 4; ++i)
    WT[(bn + ty + i * 8) * D_ + bk + tx] = (_Float16)tile[tx][ty + i * 8];
}

// ---------------------------------------------------------------------------
// GEMM: out[4096x1024] = A[4096x1024](f16) @ BT[1024x1024](f16, N-major) + bias
// mode 0: write f16, head-split layout [B,H,S,HD]  (QKV projections)
// mode 1: write f32, + residual x                  (output projection)
// Workgroup tile 128x128; 8 waves (4x2), each wave 32x64 (2x4 WMMA tiles).
// Tiles staged global->LDS via GLOBAL_LOAD_ASYNC_TO_LDS_B128, double buffered.
// ---------------------------------------------------------------------------
#define GLDA 40   // LDS row stride in halves (80B, 16B-aligned rows)

__global__ __launch_bounds__(256) void gemm128_kernel(
    const _Float16* __restrict__ A, const _Float16* __restrict__ BT,
    const float* __restrict__ bias, _Float16* __restrict__ outH,
    float* __restrict__ outF, const float* __restrict__ resid, int mode) {
  __shared__ _Float16 As[2][128 * GLDA];
  __shared__ _Float16 Bs[2][128 * GLDA];

  const int tid  = threadIdx.x;
  const int wave = tid >> 5, lane = tid & 31;
  const int wm = wave & 3, wn = wave >> 2;
  const int rowBase = blockIdx.x * 128;
  const int colBase = blockIdx.y * 128;
  const int K = D_;

  v8f acc[2][4] = {};

  const int lr   = tid >> 1;         // 0..127: tile row loaded by this thread
  const int lseg = (tid & 1) * 16;   // half-offset within 32-wide K slab

  auto stage = [&](int k0, int buf) {
    async_copy32(lds_off(&As[buf][lr * GLDA + lseg]),
                 A + (size_t)(rowBase + lr) * K + k0 + lseg);
    async_copy32(lds_off(&Bs[buf][lr * GLDA + lseg]),
                 BT + (size_t)(colBase + lr) * K + k0 + lseg);
  };

  stage(0, 0);
  asm volatile("s_wait_asynccnt 0" ::: "memory");
  __syncthreads();

  for (int k0 = 0; k0 < K; k0 += 32) {
    const int buf = (k0 >> 5) & 1;
    if (k0 + 32 < K) stage(k0 + 32, buf ^ 1);   // prefetch next slab into other buffer

    v16h af[2], bf[4];
#pragma unroll
    for (int i = 0; i < 2; ++i)
      af[i] = ldfrag(&As[buf][(wm * 32 + i * 16) * GLDA], GLDA, lane);
#pragma unroll
    for (int j = 0; j < 4; ++j)
      bf[j] = ldfrag(&Bs[buf][(wn * 64 + j * 16) * GLDA], GLDA, lane);
#pragma unroll
    for (int i = 0; i < 2; ++i)
#pragma unroll
      for (int j = 0; j < 4; ++j)
        acc[i][j] = wmma_f16(af[i], bf[j], acc[i][j]);

    asm volatile("s_wait_asynccnt 0" ::: "memory");  // next buffer filled
    __syncthreads();                                  // all waves done with buf
  }

  // Epilogue. C layout: elem e -> row = tile_m + e + 8*(lane>>4), col = tile_n + (lane&15)
  const int hf = lane >> 4, ln = lane & 15;
#pragma unroll
  for (int i = 0; i < 2; ++i) {
#pragma unroll
    for (int j = 0; j < 4; ++j) {
      const int col = colBase + wn * 64 + j * 16 + ln;
      const float bv = bias[col];
#pragma unroll
      for (int e = 0; e < 8; ++e) {
        const int row = rowBase + wm * 32 + i * 16 + e + 8 * hf;
        const float v = acc[i][j][e] + bv;
        if (mode == 0) {
          const int bb = row >> 11, s = row & (S_ - 1);
          const int h = col >> 6, d = col & (HD_ - 1);
          outH[(((size_t)bb * H_ + h) * S_ + s) * HD_ + d] = (_Float16)v;
        } else {
          const size_t idx = (size_t)row * D_ + col;
          outF[idx] = v + resid[idx];
        }
      }
    }
  }
}

// ---------------------------------------------------------------------------
// Flash attention. Grid (S/128, B*H). 8 waves: wave owns 16 q-rows.
// Q frags held in regs (pre-scaled by 1/sqrt(HD)=0.125); 64-key K blocks are
// async-staged into LDS; V blocks transposed via registers (dim-major for the
// PV B-matrix). Online softmax with 16-lane xor-shuffle row reductions;
// P goes C-layout -> LDS -> A-layout (same-wave, guarded by s_wait_dscnt).
// ---------------------------------------------------------------------------
#define LDK 72    // LDS row stride (144B, 16B-aligned rows)

__global__ __launch_bounds__(256) void attn_kernel(
    const _Float16* __restrict__ Q, const _Float16* __restrict__ Kg,
    const _Float16* __restrict__ Vg, const int* __restrict__ mask,
    _Float16* __restrict__ ctx) {
  __shared__ _Float16 Ks[64 * LDK];
  __shared__ _Float16 VTs[64 * LDK];
  __shared__ _Float16 Ps[8 * 16 * LDK];

  const int tid  = threadIdx.x;
  const int wave = tid >> 5, lane = tid & 31;
  const int hf = lane >> 4, ln = lane & 15;
  const int bh = blockIdx.y;                 // 0..31
  const int b = bh >> 4, h = bh & (H_ - 1);
  const size_t base = (size_t)bh * S_ * HD_;
  const int qRow = blockIdx.x * 128 + wave * 16;   // s-index of wave's first row

  // Q fragments, pre-scaled by NORM = 0.125 (exact in f16)
  v16h qf[2];
#pragma unroll
  for (int kk = 0; kk < 2; ++kk) {
    qf[kk] = ldfrag(Q + base + (size_t)qRow * HD_ + kk * 32, HD_, lane);
    qf[kk] = qf[kk] * (_Float16)0.125f;
  }

  float mrow[8], lrow[8];
  v8f o[4] = {};
#pragma unroll
  for (int e = 0; e < 8; ++e) { mrow[e] = -1e30f; lrow[e] = 0.f; }

  const int ldr  = tid >> 2;        // 0..63: key row loaded by this thread
  const int lds4 = (tid & 3) * 16;  // 16-half segment
  _Float16* pw = Ps + wave * 16 * LDK;
  const unsigned ldsK = lds_off(&Ks[ldr * LDK + lds4]);

  for (int k0 = 0; k0 < S_; k0 += 64) {
    // ---- V into registers (transposed store below); K async to LDS
    const uint4* gv = (const uint4*)(Vg + base + (size_t)(k0 + ldr) * HD_ + lds4);
    uint4 vv0 = gv[0], vv1 = gv[1];
    __syncthreads();                 // previous iteration done with LDS
    async_copy32(ldsK, Kg + base + (size_t)(k0 + ldr) * HD_ + lds4);
    {
      const _Float16* vh = (const _Float16*)&vv0;
#pragma unroll
      for (int i = 0; i < 8; ++i) VTs[(lds4 + i) * LDK + ldr] = vh[i];
      vh = (const _Float16*)&vv1;
#pragma unroll
      for (int i = 0; i < 8; ++i) VTs[(lds4 + 8 + i) * LDK + ldr] = vh[i];
    }
    asm volatile("s_wait_asynccnt 0" ::: "memory");
    __syncthreads();

    // ---- S = (Q*NORM) @ K^T   (16x64 per wave, 4 N-tiles, 2 K-steps)
    v8f sc[4] = {};
#pragma unroll
    for (int kk = 0; kk < 2; ++kk)
#pragma unroll
      for (int n = 0; n < 4; ++n)
        sc[n] = wmma_f16(qf[kk], ldfrag(&Ks[(n * 16) * LDK + kk * 32], LDK, lane),
                         sc[n]);

    // ---- mask (scores -> -1e9 where mask==0)
#pragma unroll
    for (int n = 0; n < 4; ++n) {
      const int col = k0 + n * 16 + ln;
#pragma unroll
      for (int e = 0; e < 8; ++e) {
        const int row = qRow + e + 8 * hf;
        if (mask[(size_t)row * S_ + col] == 0) sc[n][e] = -1e9f;
      }
    }

    // ---- online softmax per row (rows replicated across each 16-lane half)
#pragma unroll
    for (int e = 0; e < 8; ++e) {
      float mx = fmaxf(fmaxf(sc[0][e], sc[1][e]), fmaxf(sc[2][e], sc[3][e]));
      mx = fmaxf(mx, __shfl_xor(mx, 1));
      mx = fmaxf(mx, __shfl_xor(mx, 2));
      mx = fmaxf(mx, __shfl_xor(mx, 4));
      mx = fmaxf(mx, __shfl_xor(mx, 8));
      const float mnew = fmaxf(mrow[e], mx);
      const float corr = __expf(mrow[e] - mnew);
      float psum = 0.f;
#pragma unroll
      for (int n = 0; n < 4; ++n) {
        const float p = __expf(sc[n][e] - mnew);
        sc[n][e] = p;
        psum += p;
      }
      psum += __shfl_xor(psum, 1);
      psum += __shfl_xor(psum, 2);
      psum += __shfl_xor(psum, 4);
      psum += __shfl_xor(psum, 8);
      lrow[e] = lrow[e] * corr + psum;
      mrow[e] = mnew;
#pragma unroll
      for (int n = 0; n < 4; ++n) o[n][e] *= corr;
      // P: C-layout -> per-wave LDS strip (row e+8*hf, col n*16+ln)
#pragma unroll
      for (int n = 0; n < 4; ++n)
        pw[(e + 8 * hf) * LDK + n * 16 + ln] = (_Float16)sc[n][e];
    }

    // same-wave LDS RAW across lanes: drain DS counter before fragment reads
    asm volatile("s_wait_dscnt 0" ::: "memory");

    // ---- O += P @ V   (A = P strip, B = V^T dim-major; 2 K-steps over keys)
#pragma unroll
    for (int kk = 0; kk < 2; ++kk) {
      v16h pf = ldfrag(pw + kk * 32, LDK, lane);
#pragma unroll
      for (int n = 0; n < 4; ++n)
        o[n] = wmma_f16(pf, ldfrag(&VTs[(n * 16) * LDK + kk * 32], LDK, lane),
                        o[n]);
    }
  }

  // ---- epilogue: O /= l, write ctx in [B,S,D] with head-major columns
#pragma unroll
  for (int e = 0; e < 8; ++e) {
    const float inv = 1.f / lrow[e];
    const int row = qRow + e + 8 * hf;
#pragma unroll
    for (int n = 0; n < 4; ++n) {
      const int d = n * 16 + ln;
      ctx[((size_t)(b * S_ + row)) * D_ + h * HD_ + d] = (_Float16)(o[n][e] * inv);
    }
  }
}

// ---------------------------------------------------------------------------
// LayerNorm over D=1024, one block per row. y already contains out+residual.
// ---------------------------------------------------------------------------
__global__ __launch_bounds__(256) void layernorm_kernel(
    const float* __restrict__ y, const float* __restrict__ gamma,
    const float* __restrict__ beta, float* __restrict__ out) {
  __shared__ float ws1[8], ws2[8];
  const int tid = threadIdx.x, wave = tid >> 5, lane = tid & 31;
  const size_t row = blockIdx.x;
  const float* yr = y + row * D_;

  float v[4], s = 0.f, s2 = 0.f;
#pragma unroll
  for (int i = 0; i < 4; ++i) {
    v[i] = yr[tid + i * 256];
    s += v[i];
    s2 += v[i] * v[i];
  }
#pragma unroll
  for (int m = 1; m < 32; m <<= 1) {
    s += __shfl_xor(s, m);
    s2 += __shfl_xor(s2, m);
  }
  if (lane == 0) { ws1[wave] = s; ws2[wave] = s2; }
  __syncthreads();
  s = 0.f; s2 = 0.f;
#pragma unroll
  for (int w = 0; w < 8; ++w) { s += ws1[w]; s2 += ws2[w]; }
  const float mu = s * (1.f / D_);
  const float var = s2 * (1.f / D_) - mu * mu;
  const float r = rsqrtf(var + 1e-5f);
#pragma unroll
  for (int i = 0; i < 4; ++i) {
    const int c = tid + i * 256;
    out[row * D_ + c] = (v[i] - mu) * r * gamma[c] + beta[c];
  }
}

// ---------------------------------------------------------------------------
// Launch. Workspace layout (64 MB):
//  [0,8M)   xh f16            [8M,16M)  WqT/WkT/WvT/WoT f16 (2MB each)
//  [16M,24M) Q  [24M,32M) K  [32M,40M) V   (f16, [B,H,S,HD])
//  [40M,48M) ctx f16 [B,S,D]  [48M,64M) y f32 (proj+bias+residual)
// ---------------------------------------------------------------------------
extern "C" void kernel_launch(void* const* d_in, const int* in_sizes, int n_in,
                              void* d_out, int out_size, void* d_ws, size_t ws_size,
                              hipStream_t stream) {
  const float* x    = (const float*)d_in[0];
  const int*   mask = (const int*)d_in[1];
  const float* Wq = (const float*)d_in[2];  const float* bq = (const float*)d_in[3];
  const float* Wk = (const float*)d_in[4];  const float* bk = (const float*)d_in[5];
  const float* Wv = (const float*)d_in[6];  const float* bv = (const float*)d_in[7];
  const float* Wo = (const float*)d_in[8];  const float* bo = (const float*)d_in[9];
  const float* gamma = (const float*)d_in[10];
  const float* beta  = (const float*)d_in[11];

  char* ws = (char*)d_ws;
  _Float16* xh  = (_Float16*)(ws);
  _Float16* WqT = (_Float16*)(ws + (8ull << 20));
  _Float16* WkT = (_Float16*)(ws + (10ull << 20));
  _Float16* WvT = (_Float16*)(ws + (12ull << 20));
  _Float16* WoT = (_Float16*)(ws + (14ull << 20));
  _Float16* Qb  = (_Float16*)(ws + (16ull << 20));
  _Float16* Kb  = (_Float16*)(ws + (24ull << 20));
  _Float16* Vb  = (_Float16*)(ws + (32ull << 20));
  _Float16* ctx = (_Float16*)(ws + (40ull << 20));
  float*    yb  = (float*)(ws + (48ull << 20));

  // x -> f16 (4M elems, 4/thread)
  cvt_f16_kernel<<<4096, 256, 0, stream>>>(x, xh);

  // weights -> f16, transposed to N-major
  dim3 tb(32, 8), tg(32, 32);
  transpose_cvt_kernel<<<tg, tb, 0, stream>>>(Wq, WqT);
  transpose_cvt_kernel<<<tg, tb, 0, stream>>>(Wk, WkT);
  transpose_cvt_kernel<<<tg, tb, 0, stream>>>(Wv, WvT);
  transpose_cvt_kernel<<<tg, tb, 0, stream>>>(Wo, WoT);

  // QKV projections (head-split f16 outputs)
  dim3 gg(32, 8);   // (4096/128 rows, 1024/128 cols)
  gemm128_kernel<<<gg, 256, 0, stream>>>(xh, WqT, bq, Qb, nullptr, nullptr, 0);
  gemm128_kernel<<<gg, 256, 0, stream>>>(xh, WkT, bk, Kb, nullptr, nullptr, 0);
  gemm128_kernel<<<gg, 256, 0, stream>>>(xh, WvT, bv, Vb, nullptr, nullptr, 0);

  // attention
  attn_kernel<<<dim3(S_ / 128, B_ * H_), 256, 0, stream>>>(Qb, Kb, Vb, mask, ctx);

  // output projection + bias + residual (fp32)
  gemm128_kernel<<<gg, 256, 0, stream>>>(ctx, WoT, bo, nullptr, yb, x, 1);

  // LayerNorm -> d_out
  layernorm_kernel<<<B_ * S_, 256, 0, stream>>>(yb, gamma, beta, (float*)d_out);
}
